// Net_13340168421477
// MI455X (gfx1250) — matrix operand order repaired and checked by reading, not compiled, not run on hardware
//
#include <hip/hip_runtime.h>
#include <hip/hip_bf16.h>

#define NNODES 100000
#define FIN    501
#define KP1    512      // FIN padded to multiple of 32
#define LDSTR  (KP1 + 8) // LDS row stride in ushorts (1040B: 16B-aligned, conflict-mitigating)
#define H1     8
#define C1     8
#define HC1    64       // H1*C1
#define NCLS   40
#define NCLSP  48       // NCLS padded to multiple of 16
#define NEGS   0.2f

typedef __attribute__((ext_vector_type(16))) __bf16 v16bf;
typedef __attribute__((ext_vector_type(8)))  float  v8f;

// ---------- helpers ----------
__device__ __forceinline__ unsigned short f32_to_bf16_bits(float f) {
  unsigned int u = __float_as_uint(f);
  unsigned int r = u + 0x7FFFu + ((u >> 16) & 1u);   // round-to-nearest-even
  return (unsigned short)(r >> 16);
}
__device__ __forceinline__ __bf16 bf16_from_f32(float f) {
  union { unsigned short s; __bf16 b; } cv;
  cv.s = f32_to_bf16_bits(f);
  return cv.b;
}
// order-preserving float -> uint mapping for atomic max
__device__ __forceinline__ unsigned int fkey(float f) {
  unsigned int u = __float_as_uint(f);
  return (u & 0x80000000u) ? ~u : (u | 0x80000000u);
}
__device__ __forceinline__ float funkey(unsigned int k) {
  unsigned int u = (k & 0x80000000u) ? (k & 0x7FFFFFFFu) : ~k;
  return __uint_as_float(u);
}
__device__ __forceinline__ void edge_sd(long long e, const int* __restrict__ src,
                                        const int* __restrict__ dst, int E,
                                        int* s, int* d) {
  if (e < (long long)E) { *s = src[e]; *d = dst[e]; }
  else { *s = *d = (int)(e - E); }   // self loop
}

// ---------- weight transpose + pad to bf16 ----------
__global__ void k_wconv1(const float* __restrict__ W1, unsigned short* __restrict__ w1t) {
  int t = blockIdx.x * blockDim.x + threadIdx.x;     // [HC1][KP1]
  if (t >= HC1 * KP1) return;
  int n = t / KP1, k = t % KP1;
  float v = (k < FIN) ? W1[(long long)k * HC1 + n] : 0.0f;
  w1t[t] = f32_to_bf16_bits(v);
}
__global__ void k_wconv2(const float* __restrict__ W2, unsigned short* __restrict__ w2t) {
  int t = blockIdx.x * blockDim.x + threadIdx.x;     // [NCLSP][HC1]
  if (t >= NCLSP * HC1) return;
  int n = t / HC1, k = t % HC1;
  float v = (n < NCLS) ? W2[(long long)k * NCLS + n] : 0.0f;
  w2t[t] = f32_to_bf16_bits(v);
}

// ---------- init ----------
__global__ void k_init_seg(unsigned int* __restrict__ m, float* __restrict__ den, long long n) {
  long long t = (long long)blockIdx.x * blockDim.x + threadIdx.x;
  if (t < n) { m[t] = 0u; den[t] = 0.0f; }           // fkey domain minimum is 0
}
__global__ void k_zero(float* __restrict__ p, long long n) {
  long long t = (long long)blockIdx.x * blockDim.x + threadIdx.x;
  if (t < n) p[t] = 0.0f;
}
__global__ void k_init_out(float* __restrict__ out, const float* __restrict__ bias) {
  long long t = (long long)blockIdx.x * blockDim.x + threadIdx.x;
  if (t < (long long)NNODES * NCLS) out[t] = bias[t % NCLS];
}

// ---------- layer-1 GEMM: xh1 = x @ W1  (bf16 WMMA, f32 accum, LDS-staged A) ----------
// block = 128 threads (4 waves); wave w -> output cols [16w,16w+16); grid.x = NNODES/16
__global__ void k_gemm1(const float* __restrict__ x, const unsigned short* __restrict__ w1t,
                        float* __restrict__ xh1) {
  __shared__ unsigned short As[16 * LDSTR];          // 16 rows x 512 bf16 (+pad)

  const int tid = threadIdx.x;
  const long long rowbase = (long long)blockIdx.x * 16;

  // Cooperative, coalesced load of the 16x501 f32 strip -> bf16 LDS tile.
  // Flat index: consecutive threads hit consecutive K within a row.
  for (int it = tid; it < 16 * KP1; it += 128) {
    int rr = it >> 9;                                // / KP1
    int kk = it & (KP1 - 1);
    float v = (kk < FIN) ? x[(rowbase + rr) * FIN + kk] : 0.0f;
    As[rr * LDSTR + kk] = f32_to_bf16_bits(v);
  }
  __syncthreads();

  const int lane = tid & 31;
  const int wave = tid >> 5;
  const int r    = lane & 15;
  const int kh   = lane >> 4;                        // K-half select
  const int ncol = wave * 16 + r;
  const unsigned short* arow = &As[r * LDSTR];
  const unsigned short* wrow = w1t + (long long)ncol * KP1 + kh * 8;

  v8f c = {};
#pragma unroll 4
  for (int kb = 0; kb < KP1; kb += 32) {
    // A fragment: lane half 0 -> K {kb..kb+7, kb+16..kb+23}; half 1 -> +8
    union { uint4 q[2]; v16bf v; } ua;
    ua.q[0] = *reinterpret_cast<const uint4*>(arow + kb + kh * 8);        // ds_load_b128
    ua.q[1] = *reinterpret_cast<const uint4*>(arow + kb + 16 + kh * 8);   // ds_load_b128
    // B fragment: column `ncol`, K {kb..kb+7, kb+16..kb+23} + kh*8 (matches A)
    union { uint4 q[2]; v16bf v; } ub;
    ub.q[0] = *reinterpret_cast<const uint4*>(wrow + kb);
    ub.q[1] = *reinterpret_cast<const uint4*>(wrow + kb + 16);
    c = __builtin_amdgcn_wmma_f32_16x16x32_bf16(false, ua.v, false, ub.v,
                                                (short)0, c, false, false);
  }
#pragma unroll
  for (int j = 0; j < 8; ++j)
    xh1[(rowbase + kh * 8 + j) * HC1 + ncol] = c[j];
}

// ---------- layer-2 GEMM: xh2 = h @ W2 ----------
// block = 96 threads (3 waves -> 48 padded cols); grid.x = NNODES/16
__global__ void k_gemm2(const float* __restrict__ h, const unsigned short* __restrict__ w2t,
                        float* __restrict__ xh2) {
  const int lane = threadIdx.x & 31;
  const int wave = threadIdx.x >> 5;
  const int r    = lane & 15;
  const int kh   = lane >> 4;
  const long long gm = (long long)blockIdx.x * 16 + r;
  const int ncol = wave * 16 + r;
  const float* hrow = h + gm * HC1;                  // 256B-aligned rows
  const unsigned short* wrow = w2t + (long long)ncol * HC1 + kh * 8;
  v8f c = {};
#pragma unroll
  for (int kb = 0; kb < HC1; kb += 32) {
    const float4* p1 = reinterpret_cast<const float4*>(hrow + kb + kh * 8);       // aligned b128
    const float4* p2 = reinterpret_cast<const float4*>(hrow + kb + 16 + kh * 8);  // aligned b128
    float4 f0 = p1[0], f1 = p1[1];
    float4 g0 = p2[0], g1 = p2[1];
    union { __bf16 e[16]; v16bf v; } ua;
    ua.e[0]  = bf16_from_f32(f0.x); ua.e[1]  = bf16_from_f32(f0.y);
    ua.e[2]  = bf16_from_f32(f0.z); ua.e[3]  = bf16_from_f32(f0.w);
    ua.e[4]  = bf16_from_f32(f1.x); ua.e[5]  = bf16_from_f32(f1.y);
    ua.e[6]  = bf16_from_f32(f1.z); ua.e[7]  = bf16_from_f32(f1.w);
    ua.e[8]  = bf16_from_f32(g0.x); ua.e[9]  = bf16_from_f32(g0.y);
    ua.e[10] = bf16_from_f32(g0.z); ua.e[11] = bf16_from_f32(g0.w);
    ua.e[12] = bf16_from_f32(g1.x); ua.e[13] = bf16_from_f32(g1.y);
    ua.e[14] = bf16_from_f32(g1.z); ua.e[15] = bf16_from_f32(g1.w);
    union { uint4 q[2]; v16bf v; } ub;
    ub.q[0] = *reinterpret_cast<const uint4*>(wrow + kb);
    ub.q[1] = *reinterpret_cast<const uint4*>(wrow + kb + 16);
    c = __builtin_amdgcn_wmma_f32_16x16x32_bf16(false, ua.v, false, ub.v,
                                                (short)0, c, false, false);
  }
#pragma unroll
  for (int j = 0; j < 8; ++j)
    if (ncol < NCLS)
      xh2[((long long)blockIdx.x * 16 + kh * 8 + j) * NCLS + ncol] = c[j];
}

// ---------- attention coefficients ----------
__global__ void k_att1(const float* __restrict__ xh1, const float* __restrict__ as,
                       const float* __restrict__ ad, const int* __restrict__ idxp,
                       float* __restrict__ asrc, float* __restrict__ adst) {
  int t = blockIdx.x * blockDim.x + threadIdx.x;     // N*H1
  if (t >= NNODES * H1) return;
  int i = t >> 3, hh = t & 7;
  float sign = (idxp[0] == 1) ? -1.0f : 1.0f;
  const float* row = xh1 + (long long)i * HC1 + hh * C1;
  float ss = 0.0f, dd = 0.0f;
#pragma unroll
  for (int c = 0; c < C1; ++c) {
    float v = row[c];
    ss += v * (sign * as[hh * C1 + c]);
    dd += v * (sign * ad[hh * C1 + c]);
  }
  asrc[t] = ss; adst[t] = dd;
}
__global__ void k_att2(const float* __restrict__ xh2, const float* __restrict__ as,
                       const float* __restrict__ ad, const int* __restrict__ idxp,
                       float* __restrict__ asrc, float* __restrict__ adst) {
  int i = blockIdx.x * blockDim.x + threadIdx.x;
  if (i >= NNODES) return;
  float sign = (idxp[0] == 1) ? -1.0f : 1.0f;
  const float* row = xh2 + (long long)i * NCLS;
  float ss = 0.0f, dd = 0.0f;
#pragma unroll
  for (int c = 0; c < NCLS; ++c) {
    float v = row[c];
    ss += v * (sign * as[c]);
    dd += v * (sign * ad[c]);
  }
  asrc[i] = ss; adst[i] = dd;
}

// ---------- layer-1 edge passes (H=8) ----------
__global__ void k_max1(const int* __restrict__ src, const int* __restrict__ dst, int E,
                       const float* __restrict__ asrc, const float* __restrict__ adst,
                       unsigned int* __restrict__ m1u) {
  long long t = (long long)blockIdx.x * blockDim.x + threadIdx.x;
  long long T = ((long long)E + NNODES) * H1;
  if (t >= T) return;
  int hh = (int)(t & 7); long long e = t >> 3;
  int s, d; edge_sd(e, src, dst, E, &s, &d);
  float al = asrc[(long long)s * H1 + hh] + adst[(long long)d * H1 + hh];
  al = al > 0.0f ? al : NEGS * al;
  atomicMax(&m1u[(long long)d * H1 + hh], fkey(al));
}
__global__ void k_sum1(const int* __restrict__ src, const int* __restrict__ dst, int E,
                       const float* __restrict__ asrc, const float* __restrict__ adst,
                       const unsigned int* __restrict__ m1u, float* __restrict__ den) {
  long long t = (long long)blockIdx.x * blockDim.x + threadIdx.x;
  long long T = ((long long)E + NNODES) * H1;
  if (t >= T) return;
  int hh = (int)(t & 7); long long e = t >> 3;
  int s, d; edge_sd(e, src, dst, E, &s, &d);
  float al = asrc[(long long)s * H1 + hh] + adst[(long long)d * H1 + hh];
  al = al > 0.0f ? al : NEGS * al;
  float ex = expf(al - funkey(m1u[(long long)d * H1 + hh]));
  atomicAdd(&den[(long long)d * H1 + hh], ex);
}
__global__ void k_aggr1(const int* __restrict__ src, const int* __restrict__ dst, int E,
                        const float* __restrict__ asrc, const float* __restrict__ adst,
                        const unsigned int* __restrict__ m1u, const float* __restrict__ den,
                        const float* __restrict__ xh1, float* __restrict__ out1) {
  long long t = (long long)blockIdx.x * blockDim.x + threadIdx.x;
  long long T = ((long long)E + NNODES) * H1;
  if (t >= T) return;
  int hh = (int)(t & 7); long long e = t >> 3;
  int s, d; edge_sd(e, src, dst, E, &s, &d);
  float al = asrc[(long long)s * H1 + hh] + adst[(long long)d * H1 + hh];
  al = al > 0.0f ? al : NEGS * al;
  float a = expf(al - funkey(m1u[(long long)d * H1 + hh])) /
            (den[(long long)d * H1 + hh] + 1e-16f);
  const float* xr = xh1 + (long long)s * HC1 + hh * C1;
  float* orow = out1 + (long long)d * HC1 + hh * C1;
#pragma unroll
  for (int c = 0; c < C1; ++c) atomicAdd(&orow[c], a * xr[c]);
}

// ---------- ELU (in place) ----------
__global__ void k_elu(float* __restrict__ h) {
  long long t = (long long)blockIdx.x * blockDim.x + threadIdx.x;
  if (t >= (long long)NNODES * HC1) return;
  float v = h[t];
  h[t] = v > 0.0f ? v : (expf(v) - 1.0f);
}

// ---------- layer-2 edge passes (H=1, C=40) ----------
__global__ void k_max2(const int* __restrict__ src, const int* __restrict__ dst, int E,
                       const float* __restrict__ asrc, const float* __restrict__ adst,
                       unsigned int* __restrict__ m2u) {
  long long e = (long long)blockIdx.x * blockDim.x + threadIdx.x;
  if (e >= (long long)E + NNODES) return;
  int s, d; edge_sd(e, src, dst, E, &s, &d);
  float al = asrc[s] + adst[d];
  al = al > 0.0f ? al : NEGS * al;
  atomicMax(&m2u[d], fkey(al));
}
__global__ void k_sum2(const int* __restrict__ src, const int* __restrict__ dst, int E,
                       const float* __restrict__ asrc, const float* __restrict__ adst,
                       const unsigned int* __restrict__ m2u, float* __restrict__ den) {
  long long e = (long long)blockIdx.x * blockDim.x + threadIdx.x;
  if (e >= (long long)E + NNODES) return;
  int s, d; edge_sd(e, src, dst, E, &s, &d);
  float al = asrc[s] + adst[d];
  al = al > 0.0f ? al : NEGS * al;
  atomicAdd(&den[d], expf(al - funkey(m2u[d])));
}
__global__ void k_aggr2(const int* __restrict__ src, const int* __restrict__ dst, int E,
                        const float* __restrict__ asrc, const float* __restrict__ adst,
                        const unsigned int* __restrict__ m2u, const float* __restrict__ den,
                        const float* __restrict__ xh2, float* __restrict__ out) {
  long long t = (long long)blockIdx.x * blockDim.x + threadIdx.x;
  long long T = ((long long)E + NNODES) * 5;         // 5 groups of 8 cols
  if (t >= T) return;
  long long e = t / 5; int g = (int)(t - e * 5);
  int s, d; edge_sd(e, src, dst, E, &s, &d);
  float al = asrc[s] + adst[d];
  al = al > 0.0f ? al : NEGS * al;
  float a = expf(al - funkey(m2u[d])) / (den[d] + 1e-16f);
  const float* xr = xh2 + (long long)s * NCLS + g * 8;
  float* orow = out + (long long)d * NCLS + g * 8;
#pragma unroll
  for (int c = 0; c < 8; ++c) atomicAdd(&orow[c], a * xr[c]);
}

// ---------- launch ----------
static inline unsigned int cdiv(long long a, long long b) { return (unsigned int)((a + b - 1) / b); }

extern "C" void kernel_launch(void* const* d_in, const int* in_sizes, int n_in,
                              void* d_out, int out_size, void* d_ws, size_t ws_size,
                              hipStream_t stream) {
  const float* x     = (const float*)d_in[0];
  const float* W1    = (const float*)d_in[1];
  const float* as1   = (const float*)d_in[2];
  const float* ad1   = (const float*)d_in[3];
  const float* W2    = (const float*)d_in[4];
  const float* as2   = (const float*)d_in[5];
  const float* ad2   = (const float*)d_in[6];
  const float* bias2 = (const float*)d_in[7];
  const int*   ei    = (const int*)d_in[8];
  const int*   idxp  = (const int*)d_in[9];
  const int E = in_sizes[8] / 2;
  const int* src = ei;
  const int* dst = ei + E;
  float* out = (float*)d_out;

  // workspace carve-up (all 16B aligned)
  char* w = (char*)d_ws;
  float* xh1 = (float*)w;                    w += (size_t)NNODES * HC1 * 4;
  float* h1  = (float*)w;                    w += (size_t)NNODES * HC1 * 4;   // aggregated layer-1 out / h
  float* asr1 = (float*)w;                   w += (size_t)NNODES * H1 * 4;
  float* adt1 = (float*)w;                   w += (size_t)NNODES * H1 * 4;
  unsigned int* m1u = (unsigned int*)w;      w += (size_t)NNODES * H1 * 4;
  float* den1 = (float*)w;                   w += (size_t)NNODES * H1 * 4;
  float* xh2 = (float*)w;                    w += (size_t)NNODES * NCLS * 4;
  float* asr2 = (float*)w;                   w += (size_t)NNODES * 4;
  float* adt2 = (float*)w;                   w += (size_t)NNODES * 4;
  unsigned int* m2u = (unsigned int*)w;      w += (size_t)NNODES * 4;
  float* den2 = (float*)w;                   w += (size_t)NNODES * 4;
  unsigned short* w1t = (unsigned short*)w;  w += (size_t)HC1 * KP1 * 2;
  unsigned short* w2t = (unsigned short*)w;  w += (size_t)NCLSP * HC1 * 2;

  const long long EP = (long long)E + NNODES;

  // weight conversion + init
  k_wconv1<<<cdiv(HC1 * KP1, 256), 256, 0, stream>>>(W1, w1t);
  k_wconv2<<<cdiv(NCLSP * HC1, 256), 256, 0, stream>>>(W2, w2t);
  k_init_seg<<<cdiv((long long)NNODES * H1, 256), 256, 0, stream>>>(m1u, den1, (long long)NNODES * H1);
  k_zero<<<cdiv((long long)NNODES * HC1, 256), 256, 0, stream>>>(h1, (long long)NNODES * HC1);
  k_init_seg<<<cdiv(NNODES, 256), 256, 0, stream>>>(m2u, den2, NNODES);
  k_init_out<<<cdiv((long long)NNODES * NCLS, 256), 256, 0, stream>>>(out, bias2);

  // layer 1
  k_gemm1<<<NNODES / 16, 128, 0, stream>>>(x, w1t, xh1);
  k_att1<<<cdiv((long long)NNODES * H1, 256), 256, 0, stream>>>(xh1, as1, ad1, idxp, asr1, adt1);
  k_max1<<<cdiv(EP * H1, 256), 256, 0, stream>>>(src, dst, E, asr1, adt1, m1u);
  k_sum1<<<cdiv(EP * H1, 256), 256, 0, stream>>>(src, dst, E, asr1, adt1, m1u, den1);
  k_aggr1<<<cdiv(EP * H1, 256), 256, 0, stream>>>(src, dst, E, asr1, adt1, m1u, den1, xh1, h1);
  k_elu<<<cdiv((long long)NNODES * HC1, 256), 256, 0, stream>>>(h1);

  // layer 2
  k_gemm2<<<NNODES / 16, 96, 0, stream>>>(h1, w2t, xh2);
  k_att2<<<cdiv(NNODES, 256), 256, 0, stream>>>(xh2, as2, ad2, idxp, asr2, adt2);
  k_max2<<<cdiv(EP, 256), 256, 0, stream>>>(src, dst, E, asr2, adt2, m2u);
  k_sum2<<<cdiv(EP, 256), 256, 0, stream>>>(src, dst, E, asr2, adt2, m2u, den2);
  k_aggr2<<<cdiv(EP * 5, 256), 256, 0, stream>>>(src, dst, E, asr2, adt2, m2u, den2, xh2, out);
}
